// Attention_Fusion_31172872634407
// MI455X (gfx1250) — compile-verified, matrix operand order checked by reading, not run on hardware
//
#include <hip/hip_runtime.h>
#include <stdint.h>

// ---------------------------------------------------------------------------
// Shapes (fixed by the reference)
// ---------------------------------------------------------------------------
#define NB    512    // batch
#define TENC  64
#define CH    512    // C == H
#define NST   26
#define NCLSV 97
#define NGENP 128    // NCLS padded to multiple of 64 (N-strip width)
#define NEMBV 256

typedef __attribute__((ext_vector_type(16))) __bf16 v16bf;
typedef __attribute__((ext_vector_type(8)))  float  v8f;

__device__ __forceinline__ unsigned short f2bf(float f) {
  unsigned u = __float_as_uint(f);
  u += 0x7FFFu + ((u >> 16) & 1u);          // round-to-nearest-even
  return (unsigned short)(u >> 16);
}
__device__ __forceinline__ float bf2f(unsigned short s) {
  return __uint_as_float(((unsigned)s) << 16);
}
// fast transcendentals: v_exp_f32 + v_rcp_f32, no IEEE division expansion
__device__ __forceinline__ float sigf(float x) {
  return __builtin_amdgcn_rcpf(1.f + __expf(-x));
}
__device__ __forceinline__ float tanhf_fast(float x) {
  return 1.f - 2.f * __builtin_amdgcn_rcpf(1.f + __expf(2.f * x));
}

union FragBF { v16bf v; uint4 q[2]; };

// Load one k-step's worth of fragments (A + 4 B tiles; 10 B128 loads).
__device__ __forceinline__ void load_frags(const uint4* __restrict__ A4,
                                           const uint4* const* __restrict__ B4,
                                           int i0, FragBF& a, FragBF b[4]) {
  a.q[0] = A4[i0]; a.q[1] = A4[i0 + 2];
#pragma unroll
  for (int j = 0; j < 4; ++j) { b[j].q[0] = B4[j][i0]; b[j].q[1] = B4[j][i0 + 2]; }
}
// Issue the 4 WMMAs for one k-step (A fragment reused across the N-strip).
__device__ __forceinline__ void do_mm(const FragBF& a, const FragBF b[4], v8f acc[4]) {
#pragma unroll
  for (int j = 0; j < 4; ++j)
    acc[j] = __builtin_amdgcn_wmma_f32_16x16x32_bf16(false, a.v, false, b[j].v,
                                                     (short)0, acc[j],
                                                     false, false);
}

// ---------------------------------------------------------------------------
// WMMA GEMM, register-blocked 16x64 per wave, 2-stage software pipeline:
//   D[M,N] = act( A[M,K](bf16) * W[N,K]^T(bf16) + bias[N] )
// K mult of 64, M mult of 16, N mult of 64.
// Operand layout per CDNA5 ISA 7.12.2 (16-bit 16x32):
//   lane = 16*grp + l : row = l ; fragment dword v holds K pair at
//   kb = (v<4 ? 2v : 16+2(v-4)) + 8*grp  ->  row dword index
//   d = k0/2 + 4*grp + (v<4 ? v : 8+(v-4))  => two contiguous uint4 loads.
// Two fragment stages are live simultaneously, so each WMMA group waits only
// on its own stage's loads while the other stage's 10 loads are in flight.
// ---------------------------------------------------------------------------
__global__ void __launch_bounds__(256)
wmma_gemm(const unsigned short* __restrict__ A, int lda,   // [M,K] bf16
          const unsigned short* __restrict__ W,            // [N,K] bf16
          const float* __restrict__ bias,                  // may be null
          float* __restrict__ Dout,
          unsigned short* __restrict__ DoutBf,
          int ldd, int M, int N, int K, int nvalid, int act, int outbf)
{
  const int n64   = N >> 6;                       // 64-wide column strips
  const int tiles = (M >> 4) * n64;
  const int wid = blockIdx.x * 8 + (threadIdx.x >> 5);
  if (wid >= tiles) return;                       // whole-wave uniform exit
  const int lane = threadIdx.x & 31;
  const int ml   = lane & 15;
  const int grp  = lane >> 4;
  const int row0 = (wid / n64) << 4;
  const int col0 = (wid % n64) << 6;

  v8f acc[4] = {{}, {}, {}, {}};
  const uint4* A4 = (const uint4*)(A + (size_t)(row0 + ml) * lda);
  const uint4* B4[4];
#pragma unroll
  for (int j = 0; j < 4; ++j)
    B4[j] = (const uint4*)(W + (size_t)(col0 + 16 * j + ml) * K);

  FragBF a0, b0[4], a1, b1[4];
  load_frags(A4, B4, grp, a0, b0);                // k = 0
  for (int k0 = 0; k0 + 64 < K; k0 += 64) {
    load_frags(A4, B4, ((k0 + 32) >> 3) + grp, a1, b1);   // prefetch stage 1
    do_mm(a0, b0, acc);                                   // consume stage 0
    load_frags(A4, B4, ((k0 + 64) >> 3) + grp, a0, b0);   // prefetch stage 0
    do_mm(a1, b1, acc);                                   // consume stage 1
  }
  load_frags(A4, B4, ((K - 32) >> 3) + grp, a1, b1);      // last k-step
  do_mm(a0, b0, acc);
  do_mm(a1, b1, acc);

  // C/D layout: VGPR vi, lanes0-15 -> (M=vi, N=lane); lanes16-31 -> (M=vi+8)
#pragma unroll
  for (int j = 0; j < 4; ++j) {
    const int  nc  = col0 + 16 * j + ml;
    const bool nok = nc < nvalid;
    const float bv = (bias != nullptr && nok) ? bias[nc] : 0.f;
#pragma unroll
    for (int vi = 0; vi < 8; ++vi) {
      float val = acc[j][vi] + bv;
      if (act == 1) val = tanhf_fast(val);
      else if (act == 2) val = sigf(val);
      const int mr = row0 + vi + (grp << 3);
      if (nok) {
        if (outbf) DoutBf[(size_t)mr * ldd + nc] = f2bf(val);
        else       Dout [(size_t)mr * ldd + nc] = val;
      }
    }
  }
}

// ---------------------------------------------------------------------------
// Attention step: e = tanh(Hproj + hp) . w_score ; softmax over T_ENC ;
// context = alpha . batch_H ; X_bf = [context | seq_t] (bf16, GEMM operand).
// One block (8 waves) per batch row.
// ---------------------------------------------------------------------------
__global__ void __launch_bounds__(256)
attn_kernel(const unsigned short* __restrict__ HprojBf, // [B,64,512] bf16
            const float* __restrict__ hp,               // [B,512]
            const float* __restrict__ w_score,          // [512]
            const float* __restrict__ batch_H,          // [B,64,512] f32
            const float* __restrict__ attseq,           // [B,26,512] f32
            float* __restrict__ attnmap,                // [B,64,26] (output)
            unsigned short* __restrict__ Xbf,           // [B,1024] bf16
            int t)
{
  const int b = blockIdx.x, tid = threadIdx.x;
  const int lane = tid & 31, wv = tid >> 5;
  __shared__ float s_hp[CH], s_ws[CH], s_e[TENC], s_alpha[TENC];

  for (int i = tid; i < CH; i += 256) { s_hp[i] = hp[b * CH + i]; s_ws[i] = w_score[i]; }
  __syncthreads();

  for (int tt = wv; tt < TENC; tt += 8) {
    const unsigned short* Hrow = HprojBf + ((size_t)b * TENC + tt) * CH;
    float s = 0.f;
    for (int h = lane; h < CH; h += 32)
      s += tanhf_fast(bf2f(Hrow[h]) + s_hp[h]) * s_ws[h];
    for (int off = 16; off > 0; off >>= 1) s += __shfl_xor(s, off, 32);
    if (lane == 0) s_e[tt] = s;
  }
  __syncthreads();

  if (wv == 0) {                                  // softmax over 64 by wave 0
    float e0 = s_e[lane], e1 = s_e[lane + 32];
    float m = fmaxf(e0, e1);
    for (int off = 16; off > 0; off >>= 1) m = fmaxf(m, __shfl_xor(m, off, 32));
    float x0 = __expf(e0 - m), x1 = __expf(e1 - m);
    float sm = x0 + x1;
    for (int off = 16; off > 0; off >>= 1) sm += __shfl_xor(sm, off, 32);
    float inv = __builtin_amdgcn_rcpf(sm);
    s_alpha[lane] = x0 * inv; s_alpha[lane + 32] = x1 * inv;
  }
  __syncthreads();

  for (int tt = tid; tt < TENC; tt += 256)        // alpha -> [B,64,26] col t
    attnmap[((size_t)b * TENC + tt) * NST + t] = s_alpha[tt];

  for (int c = tid; c < CH; c += 256) {           // context + seq copy (bf16)
    float acc = 0.f;
    const float* Hb = batch_H + (size_t)b * TENC * CH + c;
#pragma unroll 4
    for (int tt = 0; tt < TENC; ++tt) acc += s_alpha[tt] * Hb[(size_t)tt * CH];
    Xbf[(size_t)b * 1024 + c]      = f2bf(acc);
    Xbf[(size_t)b * 1024 + CH + c] = f2bf(attseq[((size_t)b * NST + t) * CH + c]);
  }
}

// ---------------------------------------------------------------------------
// Gated-bimodal fusion elementwise + embedding gather into Xl (bf16)
// ---------------------------------------------------------------------------
__global__ void __launch_bounds__(256)
fusion_ew(const float* __restrict__ U,          // [B,1536] = [f1|f2|z] pre-act
          const int* __restrict__ text, const float* __restrict__ emb,
          unsigned short* __restrict__ Xlbf,    // [B,1280] = [I_char|emb|h]
          float* __restrict__ Chars, int t)
{
  const int idx = blockIdx.x * 256 + threadIdx.x;     // B*512
  const int b = idx >> 9, c = idx & 511;
  const float* Ub = U + (size_t)b * 1536;
  float z = sigf(Ub[1024 + c]);
  float I = z * tanhf_fast(Ub[c]) + (1.f - z) * tanhf_fast(Ub[512 + c]);
  Chars[((size_t)b * NST + t) * CH + c] = I;
  Xlbf[(size_t)b * 1280 + c] = f2bf(I);
  if (c < NEMBV) {
    int ch = text[b * NST + t];
    Xlbf[(size_t)b * 1280 + 512 + c] = f2bf(emb[(size_t)ch * NEMBV + c]);
  }
}

// ---------------------------------------------------------------------------
// LSTM elementwise; writes h (bf16) into h_bf, Xl h-slot, hid_bf, and
// the fp32 output hiddens.
// ---------------------------------------------------------------------------
__global__ void __launch_bounds__(256)
lstm_ew(const float* __restrict__ G,            // [B,2048] = [i|f|g|o]
        float* __restrict__ cbuf,
        unsigned short* __restrict__ hbf,       // [B,512] bf16 (next-step hp A)
        unsigned short* __restrict__ Xlbf,      // [B,1280] h-slot
        unsigned short* __restrict__ hidbf,     // [B,26,512] bf16 (probs A)
        float* __restrict__ hiddens, int t)
{
  const int idx = blockIdx.x * 256 + threadIdx.x;     // B*512
  const int b = idx >> 9, c = idx & 511;
  const float* Gb = G + (size_t)b * 2048;
  float ig = sigf(Gb[c]), fg = sigf(Gb[512 + c]);
  float gg = tanhf_fast(Gb[1024 + c]), og = sigf(Gb[1536 + c]);
  float cn = fg * cbuf[idx] + ig * gg;
  float hn = og * tanhf_fast(cn);
  cbuf[idx] = cn;
  unsigned short hb = f2bf(hn);
  hbf[idx] = hb;
  Xlbf[(size_t)b * 1280 + 768 + c] = hb;
  hidbf[((size_t)b * NST + t) * CH + c] = hb;
  hiddens[((size_t)b * NST + t) * CH + c] = hn;
}

// ---------------------------------------------------------------------------
// Setup kernels: pack fp32 weights/data into bf16 (block-matrix fusion)
// ---------------------------------------------------------------------------
__global__ void cvt_bf(const float* __restrict__ s, unsigned short* __restrict__ d, int n) {
  int i = blockIdx.x * 256 + threadIdx.x;
  if (i < n) d[i] = f2bf(s[i]);
}
__global__ void pack_wcat(const float* __restrict__ Wf1, const float* __restrict__ Wf2,
                          const float* __restrict__ Wz, unsigned short* __restrict__ d) {
  int i = blockIdx.x * 256 + threadIdx.x;
  if (i >= 1536 * 1024) return;
  int r = i >> 10, c = i & 1023;
  float v;
  if (r < 512)        v = (c < 512)  ? Wf1[r * 512 + c] : 0.f;
  else if (r < 1024)  v = (c >= 512) ? Wf2[(r - 512) * 512 + (c - 512)] : 0.f;
  else                v = Wz[(r - 1024) * 1024 + c];
  d[i] = f2bf(v);
}
__global__ void pack_wl(const float* __restrict__ Wih, const float* __restrict__ Whh,
                        unsigned short* __restrict__ d) {
  int i = blockIdx.x * 256 + threadIdx.x;
  if (i >= 2048 * 1280) return;
  int r = i / 1280, c = i % 1280;
  d[i] = f2bf((c < 768) ? Wih[r * 768 + c] : Whh[r * 512 + (c - 768)]);
}
__global__ void pack_wgen(const float* __restrict__ Wg, unsigned short* __restrict__ d) {
  int i = blockIdx.x * 256 + threadIdx.x;
  if (i >= NGENP * 512) return;
  int r = i >> 9, c = i & 511;
  d[i] = f2bf((r < NCLSV) ? Wg[r * 512 + c] : 0.f);
}
__global__ void pack_bias(const float* __restrict__ bf1, const float* __restrict__ bf2,
                          const float* __restrict__ bz, const float* __restrict__ bih,
                          const float* __restrict__ bhh,
                          float* __restrict__ bcat, float* __restrict__ bl) {
  int i = blockIdx.x * 256 + threadIdx.x;
  if (i < 1536) bcat[i] = (i < 512) ? bf1[i] : (i < 1024) ? bf2[i - 512] : bz[i - 1024];
  if (i < 2048) bl[i] = bih[i] + bhh[i];
}
__global__ void init_state(unsigned short* __restrict__ hbf, float* __restrict__ cbuf,
                           unsigned short* __restrict__ Xlbf) {
  int i = blockIdx.x * 256 + threadIdx.x;          // B*512
  int b = i >> 9, c = i & 511;
  hbf[i] = 0; cbuf[i] = 0.f;
  Xlbf[(size_t)b * 1280 + 768 + c] = 0;
}

// ---------------------------------------------------------------------------
// Host orchestration
// ---------------------------------------------------------------------------
extern "C" void kernel_launch(void* const* d_in, const int* in_sizes, int n_in,
                              void* d_out, int out_size, void* d_ws, size_t ws_size,
                              hipStream_t stream)
{
  (void)in_sizes; (void)n_in; (void)out_size; (void)ws_size;
  const float* batch_H = (const float*)d_in[0];
  const float* attseq  = (const float*)d_in[1];
  const int*   text    = (const int*)  d_in[2];
  const float* W_i2h   = (const float*)d_in[3];
  const float* W_h2h   = (const float*)d_in[4];
  const float* b_h2h   = (const float*)d_in[5];
  const float* w_score = (const float*)d_in[6];
  const float* W_ih    = (const float*)d_in[7];
  const float* W_hh    = (const float*)d_in[8];
  const float* b_ih    = (const float*)d_in[9];
  const float* b_hh    = (const float*)d_in[10];
  const float* Wf1     = (const float*)d_in[11];
  const float* bf1     = (const float*)d_in[12];
  const float* Wf2     = (const float*)d_in[13];
  const float* bf2     = (const float*)d_in[14];
  const float* Wz      = (const float*)d_in[15];
  const float* bz      = (const float*)d_in[16];
  const float* W_gen   = (const float*)d_in[17];
  const float* b_gen   = (const float*)d_in[18];
  const float* emb     = (const float*)d_in[19];

  float* out     = (float*)d_out;
  float* probs   = out;                                     // [B,26,97]
  float* attnmap = probs + (size_t)NB * NST * NCLSV;        // [B,64,26]
  float* hiddens = attnmap + (size_t)NB * TENC * NST;       // [B,26,512]
  float* Chars   = hiddens + (size_t)NB * NST * CH;         // [B,26,512]

  char* ws = (char*)d_ws;
  size_t o = 0;
  auto alloc = [&](size_t bytes) { size_t r = o; o += (bytes + 255) & ~(size_t)255; return r; };
  unsigned short* Wh2h_bf  = (unsigned short*)(ws + alloc((size_t)512 * 512 * 2));
  unsigned short* Wi2h_bf  = (unsigned short*)(ws + alloc((size_t)512 * 512 * 2));
  unsigned short* Wcat_bf  = (unsigned short*)(ws + alloc((size_t)1536 * 1024 * 2));
  unsigned short* Wl_bf    = (unsigned short*)(ws + alloc((size_t)2048 * 1280 * 2));
  unsigned short* Wgen_bf  = (unsigned short*)(ws + alloc((size_t)NGENP * 512 * 2));
  float*          bcat     = (float*)(ws + alloc(1536 * 4));
  float*          bl       = (float*)(ws + alloc(2048 * 4));
  unsigned short* BH_bf    = (unsigned short*)(ws + alloc((size_t)NB * TENC * CH * 2));
  unsigned short* Hproj_bf = (unsigned short*)(ws + alloc((size_t)NB * TENC * CH * 2));
  unsigned short* hid_bf   = (unsigned short*)(ws + alloc((size_t)NB * NST * CH * 2));
  unsigned short* X_bf     = (unsigned short*)(ws + alloc((size_t)NB * 1024 * 2));
  unsigned short* Xl_bf    = (unsigned short*)(ws + alloc((size_t)NB * 1280 * 2));
  unsigned short* h_bf     = (unsigned short*)(ws + alloc((size_t)NB * CH * 2));
  float*          hp       = (float*)(ws + alloc((size_t)NB * CH * 4));
  float*          U        = (float*)(ws + alloc((size_t)NB * 1536 * 4));
  float*          G        = (float*)(ws + alloc((size_t)NB * 2048 * 4));
  float*          cbuf     = (float*)(ws + alloc((size_t)NB * CH * 4));

  // ---- setup: bf16 packing + state init ---------------------------------
  cvt_bf<<<1024, 256, 0, stream>>>(W_h2h, Wh2h_bf, 512 * 512);
  cvt_bf<<<1024, 256, 0, stream>>>(W_i2h, Wi2h_bf, 512 * 512);
  cvt_bf<<<65536, 256, 0, stream>>>(batch_H, BH_bf, NB * TENC * CH);
  pack_wcat<<<6144, 256, 0, stream>>>(Wf1, Wf2, Wz, Wcat_bf);
  pack_wl<<<10240, 256, 0, stream>>>(W_ih, W_hh, Wl_bf);
  pack_wgen<<<256, 256, 0, stream>>>(W_gen, Wgen_bf);
  pack_bias<<<8, 256, 0, stream>>>(bf1, bf2, bz, b_ih, b_hh, bcat, bl);
  init_state<<<1024, 256, 0, stream>>>(h_bf, cbuf, Xl_bf);

  auto gemm = [&](const unsigned short* A, int lda, const unsigned short* W,
                  const float* bias, float* D, unsigned short* Dbf, int ldd,
                  int M, int N, int K, int nvalid, int act, int outbf) {
    int tiles = (M / 16) * (N / 64);               // 16x64 strip per wave
    wmma_gemm<<<(tiles + 7) / 8, 256, 0, stream>>>(A, lda, W, bias, D, Dbf, ldd,
                                                   M, N, K, nvalid, act, outbf);
  };

  // ---- H_proj = batch_H @ W_i2h^T (once; bf16, L2-resident) -------------
  gemm(BH_bf, CH, Wi2h_bf, nullptr, nullptr, Hproj_bf, CH,
       NB * TENC, CH, CH, CH, 0, 1);

  // ---- 26-step recurrence ----------------------------------------------
  for (int t = 0; t < NST; ++t) {
    // hp = h @ W_h2h^T + b_h2h
    gemm(h_bf, CH, Wh2h_bf, b_h2h, hp, nullptr, CH, NB, CH, CH, CH, 0, 0);
    // attention -> X_bf = [ctx | seq_t], alpha -> attnmap
    attn_kernel<<<NB, 256, 0, stream>>>(Hproj_bf, hp, w_score, batch_H, attseq,
                                        attnmap, X_bf, t);
    // U = X @ Wcat^T + bcat   (f1|f2|z pre-activation, fused block GEMM)
    gemm(X_bf, 1024, Wcat_bf, bcat, U, nullptr, 1536, NB, 1536, 1024, 1536, 0, 0);
    // I_char + embedding -> Xl_bf[:, :768], Char output
    fusion_ew<<<NB * CH / 256, 256, 0, stream>>>(U, text, emb, Xl_bf, Chars, t);
    // G = Xl @ [W_ih|W_hh]^T + (b_ih+b_hh)
    gemm(Xl_bf, 1280, Wl_bf, bl, G, nullptr, 2048, NB, 2048, 1280, 2048, 0, 0);
    // LSTM elementwise -> cbuf, h_bf, Xl h-slot, hid_bf, hiddens output
    lstm_ew<<<NB * CH / 256, 256, 0, stream>>>(G, cbuf, h_bf, Xl_bf, hid_bf,
                                               hiddens, t);
  }

  // ---- probs = hiddens @ W_gen^T + b_gen (N padded 97->128) -------------
  gemm(hid_bf, CH, Wgen_bf, b_gen, probs, nullptr, NCLSV,
       NB * NST, NGENP, CH, NCLSV, 0, 0);
}
